// WoundSegmentationGNN_10883447128135
// MI455X (gfx1250) — matrix operand order repaired and checked by reading, not compiled, not run on hardware
//
#include <hip/hip_runtime.h>
#include <hip/hip_bf16.h>
#include <math.h>

// ---------------------------------------------------------------------------
// WoundSegmentationGNN for MI455X (gfx1250, wave32, WMMA)
//   kNN gram tiles      : v_wmma_f32_16x16x4_f32  (K=4 fits xyz+pad exactly)
//   edge-conv / cls GEMM: v_wmma_f32_16x16x32_f16 (f32 accumulate)
//   top-k scan          : b128 LDS reads, register-resident compare loop
// ---------------------------------------------------------------------------

typedef __attribute__((ext_vector_type(2)))  float    v2f;
typedef __attribute__((ext_vector_type(8)))  float    v8f;
typedef __attribute__((ext_vector_type(16))) _Float16 v16h;

union H16 { v16h v; unsigned u[8]; _Float16 h[16]; };

#define BB  4
#define NN  8192
#define KNB 32

static __device__ __forceinline__ float leakyf(float x) { return x >= 0.f ? x : 0.2f * x; }

// ---------------------------------------------------------------------------
// Kernel 1: kNN (layout-parameterized; used for points (B,3,N) and x1 (B,N,64))
// One wave owns 32 query points. Per 16-candidate tile: two f32 WMMAs produce
// a 32x16 tile of -squared-distances; each lane streams its query row into a
// 32-entry replace-min pool in LDS. Top-32 set is order-invariant downstream.
// ---------------------------------------------------------------------------
#define KNN_WAVES 4
__global__ __launch_bounds__(128)
void knn_kernel(const float* __restrict__ coords, int batchStride, int sN, int sC,
                int* __restrict__ idx_out)
{
    __shared__ __align__(16) float s_tile[KNN_WAVES][32][16];
    __shared__ __align__(16) float s_poolD[KNN_WAVES][32][32];
    __shared__ int   s_poolI[KNN_WAVES][32][32];
    __shared__ float s_xq[KNN_WAVES][32];

    const int w  = threadIdx.x >> 5;
    const int l  = threadIdx.x & 31;
    const int lm = l & 15;
    const bool hi = (l >= 16);
    const int b  = blockIdx.y;
    const int qbase = (blockIdx.x * KNN_WAVES + w) * 32;
    const float* base = coords + (size_t)b * (size_t)batchStride;

    // Load query coords for both 16-query A tiles this lane touches.
    const int qa = qbase + lm;        // A-lo tile (queries 0..15)
    const int qb = qa + 16;           // A-hi tile (queries 16..31)
    float ax0 = base[(size_t)qa * sN + 0 * (size_t)sC];
    float ay0 = base[(size_t)qa * sN + 1 * (size_t)sC];
    float az0 = base[(size_t)qa * sN + 2 * (size_t)sC];
    float ax1 = base[(size_t)qb * sN + 0 * (size_t)sC];
    float ay1 = base[(size_t)qb * sN + 1 * (size_t)sC];
    float az1 = base[(size_t)qb * sN + 2 * (size_t)sC];

    // A fragment (f32 16x4): lanes 0-15 -> K={0,1}=x,y ; lanes 16-31 -> K={2,3}=z,0
    v2f Alo, Ahi;
    Alo.x = hi ? az0 : ax0;  Alo.y = hi ? 0.f : ay0;
    Ahi.x = hi ? az1 : ax1;  Ahi.y = hi ? 0.f : ay1;

    // ||q||^2 of this lane's own query (lane l owns query qbase+l)
    s_xq[w][l] = hi ? (ax1*ax1 + ay1*ay1 + az1*az1)
                    : (ax0*ax0 + ay0*ay0 + az0*az0);
    __syncthreads();

    // Per-lane cached ||q||^2 for the 8 M-rows this lane sees in each C tile.
    float xql[8], xqh[8];
    const int mbase = hi ? 8 : 0;
#pragma unroll
    for (int r = 0; r < 8; ++r) {
        xql[r] = s_xq[w][mbase + r];
        xqh[r] = s_xq[w][16 + mbase + r];
    }

    // Init pool (replace-min streaming top-k)
    for (int j = 0; j < 32; ++j) { s_poolD[w][l][j] = -3.4e38f; s_poolI[w][l][j] = 0; }
    float thr = -3.4e38f;
    int   apos = 0;

    const float4* rowv  = (const float4*)&s_tile[w][l][0];
    const float4* poolv = (const float4*)&s_poolD[w][l][0];

    for (int t = 0; t < NN / 16; ++t) {
        const int nb0 = t * 16;
        const int n = nb0 + lm;

        // gfx1250 prefetch of the next candidate tile (global_prefetch_b8)
        if (t + 1 < NN / 16)
            __builtin_prefetch(base + (size_t)(n + 16) * sN, 0, 3);

        float cx = base[(size_t)n * sN + 0 * (size_t)sC];
        float cy = base[(size_t)n * sN + 1 * (size_t)sC];
        float cz = base[(size_t)n * sN + 2 * (size_t)sC];
        float xc = cx*cx + cy*cy + cz*cz;

        // B fragment (f32 4x16): lane = column n; lanes 0-15 K={0,1}, 16-31 K={2,3}
        v2f Bv; Bv.x = hi ? cz : cx;  Bv.y = hi ? 0.f : cy;

        v8f clo = {}; v8f chi = {};
        clo = __builtin_amdgcn_wmma_f32_16x16x4_f32(false, Alo, false, Bv, (short)0, clo, false, false);
        chi = __builtin_amdgcn_wmma_f32_16x16x4_f32(false, Ahi, false, Bv, (short)0, chi, false, false);

#pragma unroll
        for (int r = 0; r < 8; ++r) {
            const int m = mbase + r;
            s_tile[w][m][lm]      = 2.f * clo[r] - xql[r] - xc;   // -(dist^2)
            s_tile[w][16 + m][lm] = 2.f * chi[r] - xqh[r] - xc;
        }
        __syncthreads();

        // Lane l scans its query row: 4x ds_load_b128, then register compares
        float4 t0 = rowv[0], t1 = rowv[1], t2 = rowv[2], t3 = rowv[3];
        float vals[16] = { t0.x, t0.y, t0.z, t0.w,  t1.x, t1.y, t1.z, t1.w,
                           t2.x, t2.y, t2.z, t2.w,  t3.x, t3.y, t3.z, t3.w };
#pragma unroll
        for (int j = 0; j < 16; ++j) {
            float v = vals[j];
            if (v > thr) {
                s_poolD[w][l][apos] = v;
                s_poolI[w][l][apos] = nb0 + j;
                // replace-min rescan: 8x ds_load_b128 over the 32-entry pool
                float mn = 3.4e38f; int mp = 0;
#pragma unroll
                for (int u = 0; u < 8; ++u) {
                    float4 q = poolv[u];
                    if (q.x < mn) { mn = q.x; mp = 4*u + 0; }
                    if (q.y < mn) { mn = q.y; mp = 4*u + 1; }
                    if (q.z < mn) { mn = q.z; mp = 4*u + 2; }
                    if (q.w < mn) { mn = q.w; mp = 4*u + 3; }
                }
                thr = mn; apos = mp;
            }
        }
        __syncthreads();
    }

    const int q = qbase + l;
    int* op = idx_out + ((size_t)(b * NN + q)) * KNB;
    for (int j = 0; j < 32; ++j) op[j] = s_poolI[w][l][j];
}

// ---------------------------------------------------------------------------
// Kernel 2: differential geometry (closed-form symmetric 3x3 eigenvalues)
// fused with the saliency MLP; writes the 9-channel feature x = [p, geom, sal]
// ---------------------------------------------------------------------------
__global__ void geom_sal_kernel(const float* __restrict__ points, const int* __restrict__ idx1,
                                const float* __restrict__ sw1,
                                const float* __restrict__ sg, const float* __restrict__ sb,
                                const float* __restrict__ sm, const float* __restrict__ sv,
                                const float* __restrict__ sw2, const float* __restrict__ sb2,
                                float* __restrict__ xout)
{
    const int tid = blockIdx.x * blockDim.x + threadIdx.x;
    if (tid >= BB * NN) return;
    const int b = tid / NN, n = tid % NN;
    const float* P = points + (size_t)b * 3 * NN;
    const int* id = idx1 + (size_t)tid * KNB;

    float Sx=0,Sy=0,Sz=0,Sxx=0,Syy=0,Szz=0,Sxy=0,Sxz=0,Syz=0;
    for (int j = 0; j < KNB; ++j) {
        int t = id[j];
        float x = P[t], y = P[NN + t], z = P[2 * NN + t];
        Sx += x; Sy += y; Sz += z;
        Sxx += x*x; Syy += y*y; Szz += z*z;
        Sxy += x*y; Sxz += x*z; Syz += y*z;
    }
    const float ik = 1.f / (float)KNB;
    float mx = Sx*ik, my = Sy*ik, mz = Sz*ik;
    float a00 = Sxx*ik - mx*mx, a11 = Syy*ik - my*my, a22 = Szz*ik - mz*mz;
    float a01 = Sxy*ik - mx*my, a02 = Sxz*ik - mx*mz, a12 = Syz*ik - my*mz;

    // Trigonometric eigenvalues of symmetric 3x3; l0 <= l1 <= l2
    float p1 = a01*a01 + a02*a02 + a12*a12;
    float q  = (a00 + a11 + a22) * (1.f/3.f);
    float p2 = (a00-q)*(a00-q) + (a11-q)*(a11-q) + (a22-q)*(a22-q) + 2.f*p1;
    float p  = sqrtf(fmaxf(p2, 0.f) * (1.f/6.f));
    float l0, l1, l2;
    if (p < 1e-12f) { l0 = l1 = l2 = q; }
    else {
        float ip = 1.f / p;
        float b00 = (a00-q)*ip, b11 = (a11-q)*ip, b22 = (a22-q)*ip;
        float b01 = a01*ip, b02 = a02*ip, b12 = a12*ip;
        float detB = b00*(b11*b22 - b12*b12) - b01*(b01*b22 - b12*b02) + b02*(b01*b12 - b11*b02);
        float r = fminf(fmaxf(0.5f * detB, -1.f), 1.f);
        float phi = acosf(r) * (1.f/3.f);
        float e1 = q + 2.f*p*cosf(phi);               // largest
        float e3 = q + 2.f*p*cosf(phi + 2.0943951023931953f); // smallest
        float e2 = 3.f*q - e1 - e3;
        l2 = e1; l1 = e2; l0 = e3;
    }
    const float eps = 1e-8f;
    float g0 = (l2 - l1) / (l2 + eps);          // linearity
    float g1 = (l1 - l0) / (l2 + eps);          // planarity
    float g2 = l0 / (l2 + eps);                 // sphericity
    float g3 = l0 / (l0 + l1 + l2 + eps);       // curvature
    float g4 = (l2 - l0) / (l2 + eps);          // anisotropy
    float geom[5] = { g0, g1, g2, g3, g4 };

    // saliency: 5 -> 16 (bn, leaky) -> 1, sigmoid
    float s = 0.f;
#pragma unroll
    for (int i = 0; i < 16; ++i) {
        float t = 0.f;
#pragma unroll
        for (int j = 0; j < 5; ++j) t += geom[j] * sw1[i*5 + j];
        t = (t - sm[i]) * (sg[i] * rsqrtf(sv[i] + 1e-5f)) + sb[i];
        s += leakyf(t) * sw2[i];
    }
    s += sb2[0];
    float sal = 1.f / (1.f + expf(-s));

    float* xr = xout + (size_t)tid * 9;
    xr[0] = P[n]; xr[1] = P[NN + n]; xr[2] = P[2*NN + n];
    xr[3] = g0; xr[4] = g1; xr[5] = g2; xr[6] = g3; xr[7] = g4; xr[8] = sal;
}

// ---------------------------------------------------------------------------
// Kernel 3: weight prep -> f16, K-padded rows for clean B-fragment b128 loads
// ---------------------------------------------------------------------------
__global__ void prep_weights(const float* __restrict__ c1w, const float* __restrict__ c2w,
                             const float* __restrict__ clsw,
                             _Float16* __restrict__ w1p, _Float16* __restrict__ w2p,
                             _Float16* __restrict__ wcp)
{
    const int t = blockIdx.x * blockDim.x + threadIdx.x;
    if (t < 64 * 32) {
        int o = t / 32, c = t % 32;
        w1p[t] = (_Float16)(c < 18 ? c1w[o * 18 + c] : 0.f);
    } else if (t < 64*32 + 64*128) {
        int j = t - 64*32;
        w2p[j] = (_Float16)c2w[j];
    } else if (t < 64*32 + 64*128 + 256*128) {
        int j = t - 64*32 - 64*128;
        wcp[j] = (_Float16)clsw[j];
    }
}

// ---------------------------------------------------------------------------
// Kernel 4: edge-conv. One wave per point: 32 (pair) rows x 64 out-channels via
// f16 WMMA (K padded to 32*KCH), bn + leaky, max over the 32 neighbor pairs.
// ---------------------------------------------------------------------------
#define CONV_WAVES 4
template<int CIN, int KCH>
__global__ __launch_bounds__(128)
void edge_conv_kernel(const float* __restrict__ xin, const int* __restrict__ idx,
                      const _Float16* __restrict__ wp,
                      const float* __restrict__ bg, const float* __restrict__ bb,
                      const float* __restrict__ bm, const float* __restrict__ bv,
                      float* __restrict__ xout)
{
    __shared__ __align__(16) _Float16 s_feat[CONV_WAVES][32][32 * KCH];
    __shared__ float s_part[CONV_WAVES][64];

    const int w  = threadIdx.x >> 5;
    const int l  = threadIdx.x & 31;
    const int lm = l & 15;
    const bool hi = (l >= 16);
    const int b  = blockIdx.y;
    const int p  = blockIdx.x * CONV_WAVES + w;
    const size_t rowp = (size_t)b * NN + p;

    // Build this lane's feature row (pair = neighbor l): [nb-c | c], pad to 32*KCH
    const float* cen = xin + rowp * CIN;
    const int nbid   = idx[rowp * KNB + l];
    const float* nbr = xin + ((size_t)b * NN + nbid) * CIN;
    for (int i = 0; i < CIN; ++i) {
        float c = cen[i], nv = nbr[i];
        s_feat[w][l][i]       = (_Float16)(nv - c);
        s_feat[w][l][CIN + i] = (_Float16)c;
    }
    for (int i = 2 * CIN; i < 32 * KCH; ++i) s_feat[w][l][i] = (_Float16)0.f;
    __syncthreads();

    // Preload A fragments (two 16-pair M tiles x KCH K-chunks) from LDS
    v16h afrag[2][KCH];
    const int koff0 = hi ? 8 : 0;
#pragma unroll
    for (int mt = 0; mt < 2; ++mt) {
        const unsigned* row = (const unsigned*)&s_feat[w][mt * 16 + lm][0];
#pragma unroll
        for (int kk = 0; kk < KCH; ++kk) {
            H16 t;
            const int o0 = (kk * 32 + koff0) >> 1;   // dword index
#pragma unroll
            for (int j = 0; j < 4; ++j) { t.u[j] = row[o0 + j]; t.u[4 + j] = row[o0 + 8 + j]; }
            afrag[mt][kk] = t.v;
        }
    }

    const int koffB = hi ? 8 : 0;
    for (int nt = 0; nt < 4; ++nt) {
        const int oc = nt * 16 + lm;
        const float scale = bg[oc] * rsqrtf(bv[oc] + 1e-5f);
        const float shift = bb[oc] - bm[oc] * scale;

        // B fragments from prepped f16 weights (row o, two 8-half K runs 16 apart)
        v16h bfrag[KCH];
        const unsigned* wrow = (const unsigned*)(wp + (size_t)oc * (32 * KCH));
#pragma unroll
        for (int kk = 0; kk < KCH; ++kk) {
            H16 t;
            const int o0 = (kk * 32 + koffB) >> 1;
#pragma unroll
            for (int j = 0; j < 4; ++j) { t.u[j] = wrow[o0 + j]; t.u[4 + j] = wrow[o0 + 8 + j]; }
            bfrag[kk] = t.v;
        }

#pragma unroll
        for (int mt = 0; mt < 2; ++mt) {
            v8f acc = {};
#pragma unroll
            for (int kk = 0; kk < KCH; ++kk)
                acc = __builtin_amdgcn_wmma_f32_16x16x32_f16(false, afrag[mt][kk], false, bfrag[kk],
                                                             (short)0, acc, false, false);
            float pm = -3.4e38f;
#pragma unroll
            for (int r = 0; r < 8; ++r) pm = fmaxf(pm, leakyf(acc[r] * scale + shift));
            s_part[w][mt * 32 + l] = pm;
        }
        __syncthreads();
        if (!hi) {
            float m0 = fmaxf(s_part[w][l],      s_part[w][16 + l]);
            float m1 = fmaxf(s_part[w][32 + l], s_part[w][48 + l]);
            xout[rowp * 64 + nt * 16 + l] = fmaxf(m0, m1);
        }
        __syncthreads();
    }
}

// ---------------------------------------------------------------------------
// Kernel 5: classifier. One wave per 16 points: (16x128)@(128x256) via WMMA,
// bn+leaky, then 256->2 contraction accumulated per-lane and LDS-reduced.
// ---------------------------------------------------------------------------
#define CLS_WAVES 8
__global__ __launch_bounds__(256)
void classifier_kernel(const float* __restrict__ x1, const float* __restrict__ x2,
                       const _Float16* __restrict__ wcp,
                       const float* __restrict__ bg, const float* __restrict__ bb,
                       const float* __restrict__ bm, const float* __restrict__ bv,
                       const float* __restrict__ w2, const float* __restrict__ b2,
                       float* __restrict__ out)
{
    __shared__ float s_red[CLS_WAVES][32][16];

    const int w  = threadIdx.x >> 5;
    const int l  = threadIdx.x & 31;
    const int lm = l & 15;
    const bool hi = (l >= 16);
    const int b  = blockIdx.y;
    const int pbase = (blockIdx.x * CLS_WAVES + w) * 16;
    const int p  = pbase + lm;
    const float* r1 = x1 + ((size_t)b * NN + p) * 64;
    const float* r2 = x2 + ((size_t)b * NN + p) * 64;

    // Preload A fragments: g = [x1 | x2], K=128 -> 4 chunks of 32
    const int koff0 = hi ? 8 : 0;
    v16h afrag[4];
#pragma unroll
    for (int kk = 0; kk < 4; ++kk) {
        H16 t;
#pragma unroll
        for (int e = 0; e < 16; ++e) {
            int K = kk * 32 + koff0 + (e < 8 ? e : 8 + e);   // e>=8 -> +16 offset
            float v = (K < 64) ? r1[K] : r2[K - 64];
            t.h[e] = (_Float16)v;
        }
        afrag[kk] = t.v;
    }

    float lacc[16];                    // [r][j] partial logits
#pragma unroll
    for (int e = 0; e < 16; ++e) lacc[e] = 0.f;

    const int koffB = hi ? 8 : 0;
    for (int nt = 0; nt < 16; ++nt) {
        const int n = nt * 16 + lm;
        const float scale = bg[n] * rsqrtf(bv[n] + 1e-5f);
        const float shift = bb[n] - bm[n] * scale;

        v16h bfrag[4];
        const unsigned* wrow = (const unsigned*)(wcp + (size_t)n * 128);
#pragma unroll
        for (int kk = 0; kk < 4; ++kk) {
            H16 t;
            const int o0 = (kk * 32 + koffB) >> 1;
#pragma unroll
            for (int j = 0; j < 4; ++j) { t.u[j] = wrow[o0 + j]; t.u[4 + j] = wrow[o0 + 8 + j]; }
            bfrag[kk] = t.v;
        }

        v8f acc = {};
#pragma unroll
        for (int kk = 0; kk < 4; ++kk)
            acc = __builtin_amdgcn_wmma_f32_16x16x32_f16(false, afrag[kk], false, bfrag[kk],
                                                         (short)0, acc, false, false);
        const float w20 = w2[n], w21 = w2[256 + n];
#pragma unroll
        for (int r = 0; r < 8; ++r) {
            float h = leakyf(acc[r] * scale + shift);
            lacc[r * 2 + 0] += h * w20;
            lacc[r * 2 + 1] += h * w21;
        }
    }

#pragma unroll
    for (int e = 0; e < 16; ++e) s_red[w][l][e] = lacc[e];
    __syncthreads();
    {
        const int m = l & 15, j = l >> 4;
        const int half = m >> 3, r = m & 7;
        float s = 0.f;
#pragma unroll
        for (int t = 0; t < 16; ++t) s += s_red[w][half * 16 + t][r * 2 + j];
        out[(size_t)b * 2 * NN + (size_t)j * NN + pbase + m] = s + b2[j];
    }
}

// ---------------------------------------------------------------------------
// Workspace layout (all offsets 256B aligned)
// ---------------------------------------------------------------------------
static constexpr size_t SZ_IDX  = (size_t)BB * NN * KNB * sizeof(int);     // 4 MB
static constexpr size_t OFF_IDX1 = 0;
static constexpr size_t OFF_IDX2 = OFF_IDX1 + SZ_IDX;
static constexpr size_t OFF_X    = OFF_IDX2 + SZ_IDX;
static constexpr size_t OFF_X1   = OFF_X  + (size_t)BB * NN * 9  * sizeof(float);
static constexpr size_t OFF_X2   = OFF_X1 + (size_t)BB * NN * 64 * sizeof(float);
static constexpr size_t OFF_W1P  = OFF_X2 + (size_t)BB * NN * 64 * sizeof(float);
static constexpr size_t OFF_W2P  = OFF_W1P + 64  * 32  * sizeof(_Float16);
static constexpr size_t OFF_WCP  = OFF_W2P + 64  * 128 * sizeof(_Float16);

extern "C" void kernel_launch(void* const* d_in, const int* in_sizes, int n_in,
                              void* d_out, int out_size, void* d_ws, size_t ws_size,
                              hipStream_t stream)
{
    const float* points  = (const float*)d_in[0];
    const float* sal_w1  = (const float*)d_in[1];
    const float* sal_g   = (const float*)d_in[2];
    const float* sal_b   = (const float*)d_in[3];
    const float* sal_m   = (const float*)d_in[4];
    const float* sal_v   = (const float*)d_in[5];
    const float* sal_w2  = (const float*)d_in[6];
    const float* sal_b2  = (const float*)d_in[7];
    const float* c1_w    = (const float*)d_in[8];
    const float* c1_g    = (const float*)d_in[9];
    const float* c1_b    = (const float*)d_in[10];
    const float* c1_m    = (const float*)d_in[11];
    const float* c1_v    = (const float*)d_in[12];
    const float* c2_w    = (const float*)d_in[13];
    const float* c2_g    = (const float*)d_in[14];
    const float* c2_b    = (const float*)d_in[15];
    const float* c2_m    = (const float*)d_in[16];
    const float* c2_v    = (const float*)d_in[17];
    const float* cls_w1  = (const float*)d_in[18];
    const float* cls_g   = (const float*)d_in[19];
    const float* cls_b   = (const float*)d_in[20];
    const float* cls_m   = (const float*)d_in[21];
    const float* cls_v   = (const float*)d_in[22];
    const float* cls_w2  = (const float*)d_in[23];
    const float* cls_b2  = (const float*)d_in[24];
    float* out = (float*)d_out;

    char* ws = (char*)d_ws;
    int*      idx1 = (int*)(ws + OFF_IDX1);
    int*      idx2 = (int*)(ws + OFF_IDX2);
    float*    x    = (float*)(ws + OFF_X);
    float*    x1   = (float*)(ws + OFF_X1);
    float*    x2   = (float*)(ws + OFF_X2);
    _Float16* w1p  = (_Float16*)(ws + OFF_W1P);
    _Float16* w2p  = (_Float16*)(ws + OFF_W2P);
    _Float16* wcp  = (_Float16*)(ws + OFF_WCP);

    // 0) weights -> f16 padded
    const int prepTotal = 64*32 + 64*128 + 256*128;
    prep_weights<<<(prepTotal + 255) / 256, 256, 0, stream>>>(c1_w, c2_w, cls_w1, w1p, w2p, wcp);

    // 1) kNN on raw points (B,3,N): strideN=1, strideC=N, batchStride=3N
    knn_kernel<<<dim3(NN / (32 * KNN_WAVES), BB), 32 * KNN_WAVES, 0, stream>>>(
        points, 3 * NN, 1, NN, idx1);

    // 2) geometry + saliency -> x (B,N,9)
    geom_sal_kernel<<<(BB * NN + 255) / 256, 256, 0, stream>>>(
        points, idx1, sal_w1, sal_g, sal_b, sal_m, sal_v, sal_w2, sal_b2, x);

    // 3) edge-conv1: CIN=9 (feat 18 -> pad 32, one K-chunk) -> x1 (B,N,64)
    edge_conv_kernel<9, 1><<<dim3(NN / CONV_WAVES, BB), 32 * CONV_WAVES, 0, stream>>>(
        x, idx1, w1p, c1_g, c1_b, c1_m, c1_v, x1);

    // 4) kNN on x1[...,:3] (B,N,64): strideN=64, strideC=1, batchStride=64N
    knn_kernel<<<dim3(NN / (32 * KNN_WAVES), BB), 32 * KNN_WAVES, 0, stream>>>(
        x1, NN * 64, 64, 1, idx2);

    // 5) edge-conv2: CIN=64 (feat 128 -> 4 K-chunks) -> x2 (B,N,64)
    edge_conv_kernel<64, 4><<<dim3(NN / CONV_WAVES, BB), 32 * CONV_WAVES, 0, stream>>>(
        x1, idx2, w2p, c2_g, c2_b, c2_m, c2_v, x2);

    // 6) classifier: concat(x1,x2) -> 256 (bn,leaky) -> 2, output (B,2,N)
    classifier_kernel<<<dim3(NN / (16 * CLS_WAVES), BB), 32 * CLS_WAVES, 0, stream>>>(
        x1, x2, wcp, cls_g, cls_b, cls_m, cls_v, cls_w2, cls_b2, out);
}
